// VQEMA_82781199663433
// MI455X (gfx1250) — matrix operand order repaired and checked by reading, not compiled, not run on hardware
//
#include <hip/hip_runtime.h>
#include <hip/hip_bf16.h>

// Problem constants (match reference)
constexpr int Bb   = 8;
constexpr int Cin  = 512;
constexpr int Nn   = 4096;
constexpr int Dd   = 256;
constexpr int Kk   = 2048;
constexpr float GAMMA = 0.99f;

typedef __attribute__((ext_vector_type(16))) __bf16 v16bf;
typedef __attribute__((ext_vector_type(8)))  float  v8f;
typedef __attribute__((ext_vector_type(2)))  float  v2f;

union BFrag { v16bf v; uint4 u[2]; };
union Pack8 { __bf16 h[8]; uint4 u; };

// ---------------------------------------------------------------------------
// Prep 1: zero the scatter accumulators (z_sum | n_sum, contiguous)
// ---------------------------------------------------------------------------
__global__ void k_zero(float* p, int n) {
    int i = blockIdx.x * 256 + threadIdx.x;
    if (i < n) p[i] = 0.0f;
}

// ---------------------------------------------------------------------------
// Prep 2: ebh[k,d] = bf16(-2*emb[k,d])  (exact scaling: power of two)
//         e_sq[k]  = sum_d emb[k,d]^2   (f32)
// ---------------------------------------------------------------------------
__global__ void k_prep(const float* __restrict__ emb,
                       __bf16* __restrict__ ebh,
                       float* __restrict__ e_sq) {
    const int k = blockIdx.x;
    const int d = threadIdx.x;
    float v = emb[(size_t)k * Dd + d];
    ebh[(size_t)k * Dd + d] = (__bf16)(-2.0f * v);
    __shared__ float red[256];
    red[d] = v * v;
    __syncthreads();
    for (int s = 128; s > 0; s >>= 1) {
        if (d < s) red[d] += red[d + s];
        __syncthreads();
    }
    if (d == 0) e_sq[k] = red[0];
}

// ---------------------------------------------------------------------------
// Conv GEMM (f32 WMMA 16x16x4): ze[b,d,n] = sum_c W[d,c] * z[b,c,n]
// Stored transposed ze_t[b,n,d] (f32) + zeh[b,n,d] (bf16).
// block = 128 threads (4 waves); wave w owns d-strip [w*64, w*64+64)
// ---------------------------------------------------------------------------
__global__ void k_conv(const float* __restrict__ z,
                       const float* __restrict__ W,
                       float* __restrict__ ze_t,
                       __bf16* __restrict__ zeh) {
    const int b    = blockIdx.y;
    const int n0   = blockIdx.x * 16;
    const int w    = threadIdx.x >> 5;
    const int lane = threadIdx.x & 31;
    const int r    = lane & 15;
    const int half = lane >> 4;
    const int dbase = w * 64;

    v8f acc[4] = {};
    const float* zb = z + (size_t)b * Cin * Nn + n0 + r;

    for (int c0 = 0; c0 < Cin; c0 += 4) {
        v2f bv;  // B frag (4x16): col n = r; VGPR v holds K = 2*half + v
        const float* q = zb + (size_t)(c0 + 2 * half) * Nn;
        bv.x = q[0];
        bv.y = q[Nn];
#pragma unroll
        for (int t = 0; t < 4; ++t) {
            const float2* ap =
                (const float2*)(W + (size_t)(dbase + t * 16 + r) * Cin + c0);
            float2 af = ap[half];
            v2f av; av.x = af.x; av.y = af.y;
            acc[t] = __builtin_amdgcn_wmma_f32_16x16x4_f32(
                false, av, false, bv, (short)0, acc[t], false, false);
        }
    }
    const int n = n0 + r;
#pragma unroll
    for (int t = 0; t < 4; ++t) {
        const int d0 = dbase + t * 16 + 8 * half;
        float* po = ze_t + ((size_t)(b * Nn + n)) * Dd + d0;
        Pack8 pk;
#pragma unroll
        for (int j = 0; j < 8; ++j) {
            float s = acc[t][j];
            po[j] = s;
            pk.h[j] = (__bf16)s;
        }
        *(uint4*)(zeh + ((size_t)(b * Nn + n)) * Dd + d0) = pk.u;
    }
}

// ---------------------------------------------------------------------------
// Distance + argmin (bf16 WMMA 16x16x32), LDS-staged codebook tiles:
//   score[k,n] = e_sq[k] + sum_d (-2*e[k,d]) * ze[b,d,n]
// block = 256 threads (8 waves); wave w owns 16 n-columns.
// The 16x256 bf16 e-tile for each k-step is copied Global->LDS ONCE per block
// by wave 0 using GLOBAL_LOAD_ASYNC_TO_LDS_B128 (ASYNCcnt), double-buffered so
// the copy of tile t+1 overlaps WMMA on tile t. LDS rows padded to 528B pitch
// (33x16B) so ds_load_b128 A-fragment reads are bank-conflict-free.
// ---------------------------------------------------------------------------
constexpr int TPITCH = 528;            // 512B row + 16B pad
constexpr int KT     = Kk / 16;        // 128 k-tiles

__device__ __forceinline__ void stage_tile(const __bf16* __restrict__ ebh,
                                           int kt, unsigned lds_base, int lane) {
    // wave 0, all 32 lanes: 16 rows x 512B, one b128 per lane per row
    const unsigned char* gbase =
        (const unsigned char*)ebh + (size_t)kt * 16 * (Dd * 2) + lane * 16;
#pragma unroll
    for (int i = 0; i < 16; ++i) {
        unsigned laddr = lds_base + i * TPITCH + lane * 16;
        unsigned long long g = (unsigned long long)(size_t)(gbase + i * (Dd * 2));
        asm volatile("global_load_async_to_lds_b128 %0, %1, off"
                     :: "v"(laddr), "v"(g) : "memory");
    }
}

__global__ void k_argmin(const __bf16* __restrict__ zeh,
                         const __bf16* __restrict__ ebh,
                         const float* __restrict__ e_sq_g,
                         int* __restrict__ min_ind) {
    __shared__ float s_esq[Kk];                                    // 8 KB
    __shared__ __align__(16) unsigned char s_tile[2][16 * TPITCH]; // 16.5 KB

    const int b    = blockIdx.y;
    const int w    = threadIdx.x >> 5;
    const int lane = threadIdx.x & 31;
    const int r    = lane & 15;
    const int half = lane >> 4;
    const int n0   = blockIdx.x * 128 + w * 16;

    // stage e_sq into LDS (published by the first barrier in the loop)
    for (int i = threadIdx.x; i < Kk; i += 256) s_esq[i] = e_sq_g[i];

    // Preload B fragments: row n=r, kk = half*16 .. half*16+15 (32B contig)
    v16bf bf[8];
    const __bf16* zrow = zeh + ((size_t)(b * Nn + n0 + r)) * Dd;
#pragma unroll
    for (int dt = 0; dt < 8; ++dt) {
        const uint4* p = (const uint4*)(zrow + dt * 32);
        BFrag f;
        f.u[0] = p[2 * half];
        f.u[1] = p[2 * half + 1];
        bf[dt] = f.v;
    }

    const unsigned tile0 = (unsigned)(size_t)(void*)&s_tile[0][0];
    const unsigned tile1 = (unsigned)(size_t)(void*)&s_tile[1][0];
    const bool staging_wave = (threadIdx.x < 32);

    if (staging_wave) stage_tile(ebh, 0, tile0, lane);

    float best  = __builtin_inff();
    int   bestk = 0;

    for (int kt = 0; kt < KT; ++kt) {
        // (1) all waves done with tile kt-1 (its buffer is overwritten next)
        __syncthreads();
        if (staging_wave) {
            if (kt + 1 < KT) {
                stage_tile(ebh, kt + 1, ((kt + 1) & 1) ? tile1 : tile0, lane);
                // in-order async completion: <=16 outstanding => tile kt done
                asm volatile("s_wait_asynccnt 16" ::: "memory");
            } else {
                asm volatile("s_wait_asynccnt 0" ::: "memory");
            }
        }
        // (2) publish tile kt to all waves
        __syncthreads();

        const int k0 = kt * 16;
        // C init from LDS e_sq: VGPR j holds k = k0 + 8*half + j
        const float4* eq = (const float4*)(&s_esq[k0 + 8 * half]);
        float4 e0 = eq[0], e1 = eq[1];
        v8f acc;
        acc[0] = e0.x; acc[1] = e0.y; acc[2] = e0.z; acc[3] = e0.w;
        acc[4] = e1.x; acc[5] = e1.y; acc[6] = e1.z; acc[7] = e1.w;

        const unsigned char* arow = &s_tile[kt & 1][r * TPITCH];
#pragma unroll
        for (int dt = 0; dt < 8; ++dt) {
            // A fragment from LDS: row k, kk = half*16 .. half*16+15
            const uint4* p = (const uint4*)(arow + dt * 64);
            BFrag f;
            f.u[0] = p[2 * half];
            f.u[1] = p[2 * half + 1];
            acc = __builtin_amdgcn_wmma_f32_16x16x32_bf16(
                false, f.v, false, bf[dt], (short)0, acc, false, false);
        }
#pragma unroll
        for (int j = 0; j < 8; ++j) {
            float s = acc[j];
            int  kk = k0 + 8 * half + j;
            if (s < best) { best = s; bestk = kk; }  // ascending k -> first min
        }
    }
    // Cross-half reduce (lanes L and L+16 share column n); tie -> smaller k
    float ob = __shfl_xor(best, 16, 32);
    int   ok = __shfl_xor(bestk, 16, 32);
    if (ob < best || (ob == best && ok < bestk)) { best = ob; bestk = ok; }
    if (half == 0) min_ind[b * Nn + n0 + r] = bestk;
}

// ---------------------------------------------------------------------------
// Gather: out[b,d,n] = emb[min_ind[b,n], d]   (contiguous writes along n)
// ---------------------------------------------------------------------------
__global__ void k_gather(const int* __restrict__ min_ind,
                         const float* __restrict__ emb,
                         float* __restrict__ out) {
    size_t i = (size_t)blockIdx.x * 256 + threadIdx.x;
    if (i >= (size_t)Bb * Dd * Nn) return;
    int n = (int)(i % Nn);
    size_t t = i / Nn;
    int d = (int)(t % Dd);
    int b = (int)(t / Dd);
    int k = min_ind[b * Nn + n];
    out[i] = emb[(size_t)k * Dd + d];
}

// ---------------------------------------------------------------------------
// Scatter: one 256-thread block per (b,n); contiguous f32 atomics per row
// ---------------------------------------------------------------------------
__global__ void k_scatter(const int* __restrict__ min_ind,
                          const float* __restrict__ ze_t,
                          float* __restrict__ z_sum,
                          float* __restrict__ n_sum) {
    const int bn = blockIdx.x;
    const int d  = threadIdx.x;
    const int k  = min_ind[bn];
    atomicAdd(&z_sum[(size_t)k * Dd + d], ze_t[(size_t)bn * Dd + d]);
    if (d == 0) atomicAdd(&n_sum[k], 1.0f);
}

// ---------------------------------------------------------------------------
// EMA blend into output slots
// ---------------------------------------------------------------------------
__global__ void k_ema(const float* __restrict__ ema_numer,
                      const float* __restrict__ ema_denom,
                      const float* __restrict__ z_sum,
                      const float* __restrict__ n_sum,
                      float* __restrict__ numer_out,
                      float* __restrict__ denom_out) {
    int i = blockIdx.x * 256 + threadIdx.x;
    if (i < Kk * Dd) numer_out[i] = GAMMA * ema_numer[i] + (1.0f - GAMMA) * z_sum[i];
    if (i < Kk)      denom_out[i] = GAMMA * ema_denom[i] + (1.0f - GAMMA) * n_sum[i];
}

// ---------------------------------------------------------------------------
extern "C" void kernel_launch(void* const* d_in, const int* in_sizes, int n_in,
                              void* d_out, int out_size, void* d_ws, size_t ws_size,
                              hipStream_t stream) {
    const float* z         = (const float*)d_in[0];  // (B, Cin, N)
    const float* W         = (const float*)d_in[1];  // (D, Cin)
    const float* emb       = (const float*)d_in[2];  // (K, D)
    const float* ema_numer = (const float*)d_in[3];  // (K, D)
    const float* ema_denom = (const float*)d_in[4];  // (K,)

    float* out       = (float*)d_out;                              // (B, D, N)
    float* numer_out = out + (size_t)Bb * Dd * Nn;                 // (K, D)
    float* denom_out = numer_out + (size_t)Kk * Dd;                // (K,)

    char* ws = (char*)d_ws;
    size_t off = 0;
    float*  ze_t    = (float*)(ws + off);  off += (size_t)Bb * Nn * Dd * sizeof(float);   // 32 MB
    __bf16* zeh     = (__bf16*)(ws + off); off += (size_t)Bb * Nn * Dd * sizeof(__bf16);  // 16 MB
    __bf16* ebh     = (__bf16*)(ws + off); off += (size_t)Kk * Dd * sizeof(__bf16);       //  1 MB
    float*  e_sq    = (float*)(ws + off);  off += (size_t)Kk * sizeof(float);
    float*  z_sum   = (float*)(ws + off);  off += (size_t)Kk * Dd * sizeof(float);        //  2 MB
    float*  n_sum   = (float*)(ws + off);  off += (size_t)Kk * sizeof(float);             // contiguous after z_sum
    int*    min_ind = (int*)(ws + off);    off += (size_t)Bb * Nn * sizeof(int);

    {
        int n = Kk * Dd + Kk;
        k_zero<<<(n + 255) / 256, 256, 0, stream>>>(z_sum, n);
    }
    k_prep<<<Kk, 256, 0, stream>>>(emb, ebh, e_sq);
    k_conv<<<dim3(Nn / 16, Bb), 128, 0, stream>>>(z, W, ze_t, zeh);
    k_argmin<<<dim3(Nn / 128, Bb), 256, 0, stream>>>(zeh, ebh, e_sq, min_ind);
    {
        size_t n = (size_t)Bb * Dd * Nn;
        k_gather<<<(unsigned)((n + 255) / 256), 256, 0, stream>>>(min_ind, emb, out);
    }
    k_scatter<<<Bb * Nn, 256, 0, stream>>>(min_ind, ze_t, z_sum, n_sum);
    k_ema<<<(Kk * Dd + 255) / 256, 256, 0, stream>>>(
        ema_numer, ema_denom, z_sum, n_sum, numer_out, denom_out);
}
